// MultipleInstanceLearningClassifier_15994458211325
// MI455X (gfx1250) — matrix operand order repaired and checked by reading, not compiled, not run on hardware
//
#include <hip/hip_runtime.h>
#include <cstdint>
#include <cstddef>

// ---------------------------------------------------------------------------
// CDNA5 (gfx1250) MIL classifier pipeline.
// - bf16 activations in HBM, f32 accumulation.
// - WMMA bf16 implicit-GEMM conv3/4/5 with fused maxpool+relu.
// - Async global->LDS staging (ASYNCcnt) for conv input images when available.
// - WMMA GEMMs for FC and attention-hidden layers.
// wave32 fragment layouts: contiguous k order, k = 16*(lane/16) + element (same for A and B).
// ---------------------------------------------------------------------------

typedef __attribute__((ext_vector_type(16))) __bf16 v16bf;
typedef __attribute__((ext_vector_type(8)))  float  v8f;

union FragAB {
  v16bf v;
  unsigned short u[16];
  uint4 q[2];
};
union FragC {
  v8f   v;
  float f[8];
};

__device__ __forceinline__ unsigned short f2bf(float f) {
  // round-to-nearest-even fp32 -> bf16
  unsigned int u = __float_as_uint(f);
  return (unsigned short)((u + 0x7fffu + ((u >> 16) & 1u)) >> 16);
}
__device__ __forceinline__ float bf2f(unsigned short h) {
  return __uint_as_float(((unsigned int)h) << 16);
}
__device__ __forceinline__ float loadv(const float* p, size_t i) { return p[i]; }
__device__ __forceinline__ float loadv(const unsigned short* p, size_t i) { return bf2f(p[i]); }

__device__ __forceinline__ v8f wmma_bf16(const FragAB& a, const FragAB& b, v8f c) {
  // (neg_a, A, neg_b, B, c_mod, C, reuse_a, reuse_b)
  return __builtin_amdgcn_wmma_f32_16x16x32_bf16(false, a.v, false, b.v,
                                                 (short)0, c, false, false);
}

// ---------------------------------------------------------------------------
// 16-byte global->LDS copy: async (ASYNCcnt) on gfx1250 toolchains that expose
// the builtin, synchronous fallback otherwise. Both src and dst must be 16B
// aligned. The builtin takes pointers to int4 vectors in addrspace(1)/(3);
// flat-shared low 32 bits == LDS offset (ISA aperture rules), flat global ==
// global address, so integer casts produce the addrspace pointers.
// ---------------------------------------------------------------------------
#if defined(__has_builtin)
#if __has_builtin(__builtin_amdgcn_global_load_async_to_lds_b128) && \
    __has_builtin(__builtin_amdgcn_s_wait_asynccnt)
#define USE_ASYNC_LDS 1
#endif
#endif

typedef int v4i_async __attribute__((__vector_size__(16)));

__device__ __forceinline__ void copy16_g2l(unsigned short* lds_dst,
                                           const unsigned short* g_src) {
#ifdef USE_ASYNC_LDS
  __builtin_amdgcn_global_load_async_to_lds_b128(
      (__attribute__((address_space(1))) v4i_async*)(uintptr_t)g_src,
      (__attribute__((address_space(3))) v4i_async*)(unsigned int)(uintptr_t)lds_dst,
      0, 0);
#else
  *(uint4*)lds_dst = *(const uint4*)g_src;
#endif
}
__device__ __forceinline__ void wait_async_copies() {
#ifdef USE_ASYNC_LDS
  __builtin_amdgcn_s_wait_asynccnt(0);
#endif
}

// ---------------------------------------------------------------------------
// Fragment swizzle prep kernels (-> bf16 fragments).
// A-fragment (M x K, 16x32 tiles): lane L: m = L&15;  k_local = e + (L>=16)*16
// B-fragment (K x N, 32x16 tiles): lane L: n = L&15;  k_local = e + (L>=16)*16
// ---------------------------------------------------------------------------
__global__ __launch_bounds__(256) void swizzle_a(const float* __restrict__ src,
                                                 unsigned short* __restrict__ dst,
                                                 int K, int KT, int Mtiles) {
  int gid = blockIdx.x * 256 + threadIdx.x;
  int total = Mtiles * KT * 32;
  if (gid >= total) return;
  int lane = gid & 31, frag = gid >> 5;
  int kt = frag % KT, mtile = frag / KT;
  int row = mtile * 16 + (lane & 15);
  int kofs = (lane >= 16) ? 16 : 0;
  FragAB f;
#pragma unroll
  for (int e = 0; e < 16; ++e) {
    int k = kt * 32 + e + kofs;
    f.u[e] = (k < K) ? f2bf(src[(size_t)row * K + k]) : (unsigned short)0;
  }
  uint4* o = (uint4*)(dst + (size_t)gid * 16);
  o[0] = f.q[0];
  o[1] = f.q[1];
}

__global__ __launch_bounds__(256) void swizzle_a_bf16(const unsigned short* __restrict__ src,
                                                      unsigned short* __restrict__ dst,
                                                      int K, int KT, int Mtiles) {
  int gid = blockIdx.x * 256 + threadIdx.x;
  int total = Mtiles * KT * 32;
  if (gid >= total) return;
  int lane = gid & 31, frag = gid >> 5;
  int kt = frag % KT, mtile = frag / KT;
  int row = mtile * 16 + (lane & 15);
  int kofs = (lane >= 16) ? 16 : 0;
  FragAB f;
#pragma unroll
  for (int e = 0; e < 16; ++e) {
    int k = kt * 32 + e + kofs;
    f.u[e] = (k < K) ? src[(size_t)row * K + k] : (unsigned short)0;
  }
  uint4* o = (uint4*)(dst + (size_t)gid * 16);
  o[0] = f.q[0];
  o[1] = f.q[1];
}

__global__ __launch_bounds__(256) void swizzle_b(const float* __restrict__ src, // N x K row-major
                                                 unsigned short* __restrict__ dst,
                                                 int K, int KT, int Ntiles) {
  int gid = blockIdx.x * 256 + threadIdx.x;
  int total = Ntiles * KT * 32;
  if (gid >= total) return;
  int lane = gid & 31, frag = gid >> 5;
  int kt = frag % KT, ntile = frag / KT;
  int col = ntile * 16 + (lane & 15);
  int kofs = (lane >= 16) ? 16 : 0;
  FragAB f;
#pragma unroll
  for (int e = 0; e < 16; ++e) {
    int k = kt * 32 + e + kofs;
    f.u[e] = (k < K) ? f2bf(src[(size_t)col * K + k]) : (unsigned short)0;
  }
  uint4* o = (uint4*)(dst + (size_t)gid * 16);
  o[0] = f.q[0];
  o[1] = f.q[1];
}

// ---------------------------------------------------------------------------
// Direct conv+pool+relu for the tiny-channel front layers (conv1, conv2).
// Output bf16.
// ---------------------------------------------------------------------------
template <int Ci, int Co, int Hin, typename TIN>
__global__ __launch_bounds__(256) void conv_pool_relu_direct(
    const TIN* __restrict__ in, const float* __restrict__ w,
    const float* __restrict__ bias, unsigned short* __restrict__ out) {
  constexpr int H = Hin, W = Hin;
  constexpr int Wp = H / 2, Pp = Wp * Wp;
  constexpr int CH = Pp / 256;
  __shared__ float swt[Ci * 25];
  int blk = blockIdx.x;
  int chunk = blk % CH; blk /= CH;
  int co = blk % Co;
  int b  = blk / Co;
  if ((int)threadIdx.x < Ci * 25) swt[threadIdx.x] = w[(size_t)co * Ci * 25 + threadIdx.x];
  __syncthreads();
  int q  = chunk * 256 + threadIdx.x;
  int ty = q / Wp, tx = q % Wp;
  const TIN* inb = in + (size_t)b * Ci * H * W;
  float mx = -3.4e38f;
#pragma unroll
  for (int sy = 0; sy < 2; ++sy)
#pragma unroll
    for (int sx = 0; sx < 2; ++sx) {
      int cy = 2 * ty + sy, cx = 2 * tx + sx;
      float acc = 0.f;
      for (int ci = 0; ci < Ci; ++ci)
#pragma unroll
        for (int kh = 0; kh < 5; ++kh) {
          int y = cy + kh - 2;
          if (y < 0 || y >= H) continue;
#pragma unroll
          for (int kw = 0; kw < 5; ++kw) {
            int x = cx + kw - 2;
            if (x < 0 || x >= W) continue;
            acc += loadv(inb, (size_t)(ci * H + y) * W + x) * swt[(ci * 5 + kh) * 5 + kw];
          }
        }
      mx = fmaxf(mx, acc);
    }
  out[((size_t)b * Co + co) * Pp + q] = f2bf(fmaxf(mx + bias[co], 0.f));
}

// ---------------------------------------------------------------------------
// WMMA implicit-GEMM conv + fused maxpool + relu (conv3/4/5).
// One workgroup (8 wave32) per image. bf16 input image staged in LDS via async
// global->LDS copies; left halo of 8 elements keeps 16B alignment of rows
// (row pitch WP = W+16 elems = 16B multiple). Each wave: 16 output channels x
// 16 pooled pixels; 4 accumulators = 2x2 pool sub-positions, pooled in regs.
// ---------------------------------------------------------------------------
template <int Ci, int Co, int Hin>
__global__ __launch_bounds__(256) void conv_pool_relu_wmma(
    const unsigned short* __restrict__ in, const unsigned short* __restrict__ wfrag,
    const float* __restrict__ bias, unsigned short* __restrict__ out) {
  constexpr int H = Hin, W = Hin;
  constexpr int HP = H + 4;        // 2-row halo top/bottom
  constexpr int WP = W + 16;       // 8-elem left halo, >=2 right halo, 16B pitch
  constexpr int XOFF = 8;          // interior starts 16B-aligned
  constexpr int Ktot = Ci * 25;
  constexpr int KT = (Ktot + 31) / 32;
  constexpr int Wp = H / 2;
  constexpr int Pp = Wp * Wp;
  constexpr int NG = Pp / 16;
  constexpr int Mt = Co / 16;
  constexpr int TASKS = Mt * NG;
  __shared__ alignas(16) unsigned short simg[Ci * HP * WP];

  const int b = blockIdx.x;
  const int tid = threadIdx.x;
  for (int i = tid; i < Ci * HP * WP; i += 256) simg[i] = 0;
  __syncthreads();

  // Stage interior: 16B chunks, async global->LDS.
  constexpr int XC = W / 8;               // 16B chunks per row (W in {8,16,32})
  constexpr int CHUNKS = Ci * H * XC;
  const unsigned short* inb = in + (size_t)b * Ci * H * W;
  for (int i = tid; i < CHUNKS; i += 256) {
    int ci = i / (H * XC), rem = i % (H * XC);
    int y = rem / XC, xc = rem % XC;
    copy16_g2l(&simg[(ci * HP + (y + 2)) * WP + XOFF + xc * 8],
               inb + (size_t)(ci * H + y) * W + xc * 8);
  }
  wait_async_copies();
  __syncthreads();

  const int wave = tid >> 5, lane = tid & 31;
  const bool hi = (lane >= 16);
  const int n = lane & 15;

  for (int task = wave; task < TASKS; task += 8) {
    int mtile = task % Mt, ng = task / Mt;
    int q  = ng * 16 + n;
    int ty = q / Wp, tx = q % Wp;
    int cy0 = 2 * ty, cx0 = 2 * tx;
    v8f z = {0.f, 0.f, 0.f, 0.f, 0.f, 0.f, 0.f, 0.f};
    FragC a0, a1, a2, a3;
    a0.v = z; a1.v = z; a2.v = z; a3.v = z;
    const uint4* ap = (const uint4*)(wfrag + ((size_t)(mtile * KT) * 32 + lane) * 16);
    for (int kt = 0; kt < KT; ++kt) {
      FragAB a;
      a.q[0] = ap[0];
      a.q[1] = ap[1];
      __builtin_prefetch(ap + 64, 0, 1);   // next kt's weight fragment
      ap += 64;                             // 32 lanes * 32B = 1KB per kt
      FragAB b0, b1, b2, b3;
#pragma unroll
      for (int e = 0; e < 16; ++e) {
        int k = kt * 32 + e + (hi ? 16 : 0);
        if (k >= Ktot) k = Ktot - 1;        // padded K: A=0 there, B value irrelevant
        int ci = k / 25, r = k % 25;
        int kh = r / 5, kw = r % 5;
        // input coord = (cy+kh-2, cx+kw-2); +2 row halo, +XOFF col halo
        const unsigned short* p =
            &simg[(ci * HP + (cy0 + kh)) * WP + (cx0 + kw) + (XOFF - 2)];
        b0.u[e] = p[0];
        b1.u[e] = p[1];
        b2.u[e] = p[WP];
        b3.u[e] = p[WP + 1];
      }
      a0.v = wmma_bf16(a, b0, a0.v);
      a1.v = wmma_bf16(a, b1, a1.v);
      a2.v = wmma_bf16(a, b2, a2.v);
      a3.v = wmma_bf16(a, b3, a3.v);
    }
    int mbase = mtile * 16 + (hi ? 8 : 0);
#pragma unroll
    for (int r = 0; r < 8; ++r) {
      int co = mbase + r;
      float v = fmaxf(fmaxf(a0.f[r], a1.f[r]), fmaxf(a2.f[r], a3.f[r]));
      out[((size_t)b * Co + co) * Pp + q] = f2bf(fmaxf(v + bias[co], 0.f));
    }
  }
}

// ---------------------------------------------------------------------------
// Generic WMMA GEMM on pre-swizzled fragments: out[M x N] = A * B^T + bias
// act: 0 = none, 1 = tanh.  Output fp32.
// ---------------------------------------------------------------------------
__global__ __launch_bounds__(256) void gemm_wmma(
    const unsigned short* __restrict__ Af, const unsigned short* __restrict__ Bf,
    const float* __restrict__ bias, float* __restrict__ out,
    int Ntiles, int KT, int N, int totalTiles, int act) {
  int wave = threadIdx.x >> 5, lane = threadIdx.x & 31;
  int tile = blockIdx.x * 8 + wave;
  if (tile >= totalTiles) return;
  int mtile = tile / Ntiles, ntile = tile % Ntiles;
  FragC acc;
  v8f z = {0.f, 0.f, 0.f, 0.f, 0.f, 0.f, 0.f, 0.f};
  acc.v = z;
  const uint4* ap = (const uint4*)(Af + ((size_t)(mtile * KT) * 32 + lane) * 16);
  const uint4* bp = (const uint4*)(Bf + ((size_t)(ntile * KT) * 32 + lane) * 16);
  for (int kt = 0; kt < KT; ++kt) {
    FragAB a, b;
    a.q[0] = ap[0]; a.q[1] = ap[1];
    b.q[0] = bp[0]; b.q[1] = bp[1];
    __builtin_prefetch(ap + 64, 0, 1);
    __builtin_prefetch(bp + 64, 0, 1);
    ap += 64; bp += 64;
    acc.v = wmma_bf16(a, b, acc.v);
  }
  int col = ntile * 16 + (lane & 15);
  float bv = bias[col];
  int rbase = mtile * 16 + ((lane >= 16) ? 8 : 0);
#pragma unroll
  for (int r = 0; r < 8; ++r) {
    float v = acc.f[r] + bv;
    if (act == 1) v = tanhf(v);
    out[(size_t)(rbase + r) * N + col] = v;
  }
}

// ---------------------------------------------------------------------------
// Attention scalar head: logits[i] = h[i,:] . att_w2 + att_b2
// ---------------------------------------------------------------------------
__global__ __launch_bounds__(256) void att_logits(const float* __restrict__ h,
                                                  const float* __restrict__ w2,
                                                  const float* __restrict__ b2,
                                                  float* __restrict__ logits) {
  int i = blockIdx.x * 256 + threadIdx.x;
  if (i >= 512) return;
  float s = b2[0];
  for (int a = 0; a < 128; ++a) s += h[(size_t)i * 128 + a] * w2[a];
  logits[i] = s;
}

// ---------------------------------------------------------------------------
// Segment (per-case) softmax attention pooling: one block per case.
// ---------------------------------------------------------------------------
__global__ __launch_bounds__(256) void segment_attn(const float* __restrict__ logits,
                                                    const int* __restrict__ ids,
                                                    const float* __restrict__ emb,
                                                    float* __restrict__ aggr) {
  __shared__ float wrow[512];
  __shared__ float smax, ssum;
  int c = blockIdx.x, tid = threadIdx.x;
  if (tid == 0) {
    float m = -3.4e38f;
    for (int r = 0; r < 512; ++r)
      if (ids[r] == c) m = fmaxf(m, logits[r]);
    smax = (m > -3.0e38f) ? m : 0.f;   // guard empty segments
    float s = 0.f;
    for (int r = 0; r < 512; ++r)
      if (ids[r] == c) s += expf(logits[r] - smax);
    ssum = s;
  }
  __syncthreads();
  float inv = (ssum > 0.f) ? 1.f / ssum : 0.f;
  for (int r = tid; r < 512; r += 256)
    wrow[r] = (ids[r] == c) ? expf(logits[r] - smax) * inv : 0.f;
  __syncthreads();
  for (int d = tid; d < 1024; d += 256) {
    float s = 0.f;
    for (int r = 0; r < 512; ++r) s += wrow[r] * emb[(size_t)r * 1024 + d];
    aggr[(size_t)c * 1024 + d] = s;
  }
}

__global__ void classify(const float* __restrict__ aggr, const float* __restrict__ cw,
                         const float* __restrict__ cb, float* __restrict__ out) {
  int c = blockIdx.x * blockDim.x + threadIdx.x;
  if (c >= 64) return;
  float s = cb[0];
  for (int d = 0; d < 1024; ++d) s += aggr[(size_t)c * 1024 + d] * cw[d];
  out[c] = 1.f / (1.f + expf(-s));
}

// ---------------------------------------------------------------------------
extern "C" void kernel_launch(void* const* d_in, const int* in_sizes, int n_in,
                              void* d_out, int out_size, void* d_ws, size_t ws_size,
                              hipStream_t stream) {
  (void)in_sizes; (void)n_in; (void)out_size; (void)ws_size;
  const float* data = (const float*)d_in[0];
  const int*   ids  = (const int*)d_in[1];
  const float* cw1 = (const float*)d_in[2];  const float* cb1 = (const float*)d_in[3];
  const float* cw2 = (const float*)d_in[4];  const float* cb2 = (const float*)d_in[5];
  const float* cw3 = (const float*)d_in[6];  const float* cb3 = (const float*)d_in[7];
  const float* cw4 = (const float*)d_in[8];  const float* cb4 = (const float*)d_in[9];
  const float* cw5 = (const float*)d_in[10]; const float* cb5 = (const float*)d_in[11];
  const float* fcw = (const float*)d_in[12]; const float* fcb = (const float*)d_in[13];
  const float* aw1 = (const float*)d_in[14]; const float* ab1 = (const float*)d_in[15];
  const float* aw2 = (const float*)d_in[16]; const float* ab2 = (const float*)d_in[17];
  const float* clw = (const float*)d_in[18]; const float* clb = (const float*)d_in[19];
  float* outp = (float*)d_out;

  char* base = (char*)d_ws;
  size_t off = 0;
  auto alloc = [&](size_t bytes) -> void* {
    void* p = base + off;
    off += (bytes + 255) & ~(size_t)255;
    return p;
  };
  unsigned short* p1 = (unsigned short*)alloc(512ull * 8 * 64 * 64 * 2);    // 33.6 MB
  unsigned short* p2 = (unsigned short*)alloc(512ull * 16 * 32 * 32 * 2);   // 16.8 MB
  unsigned short* p3 = (unsigned short*)alloc(512ull * 32 * 16 * 16 * 2);   // 8.4 MB
  unsigned short* p4 = (unsigned short*)alloc(512ull * 64 * 8 * 8 * 2);     // 4.2 MB
  unsigned short* p5 = (unsigned short*)alloc(512ull * 128 * 4 * 4 * 2);    // 2.1 MB
  float* emb  = (float*)alloc(512ull * 1024 * 4);
  float* hatt = (float*)alloc(512ull * 128 * 4);
  float* lgt  = (float*)alloc(512ull * 4);
  float* aggr = (float*)alloc(64ull * 1024 * 4);
  unsigned short* wsw3 = (unsigned short*)alloc(2ull  * 13 * 32 * 16 * 2);
  unsigned short* wsw4 = (unsigned short*)alloc(4ull  * 25 * 32 * 16 * 2);
  unsigned short* wsw5 = (unsigned short*)alloc(8ull  * 50 * 32 * 16 * 2);
  unsigned short* fcA  = (unsigned short*)alloc(32ull * 64 * 32 * 16 * 2);
  unsigned short* fcB  = (unsigned short*)alloc(64ull * 64 * 32 * 16 * 2);
  unsigned short* atA  = (unsigned short*)alloc(32ull * 32 * 32 * 16 * 2);
  unsigned short* atB  = (unsigned short*)alloc(8ull  * 32 * 32 * 16 * 2);

  auto gs = [](long long t) { return (int)((t + 255) / 256); };

  // Weight fragment prep (runs every launch; deterministic).
  swizzle_a<<<gs(2LL * 13 * 32), 256, 0, stream>>>(cw3, wsw3, 400, 13, 2);
  swizzle_a<<<gs(4LL * 25 * 32), 256, 0, stream>>>(cw4, wsw4, 800, 25, 4);
  swizzle_a<<<gs(8LL * 50 * 32), 256, 0, stream>>>(cw5, wsw5, 1600, 50, 8);
  swizzle_b<<<gs(64LL * 64 * 32), 256, 0, stream>>>(fcw, fcB, 2048, 64, 64);
  swizzle_b<<<gs(8LL * 32 * 32), 256, 0, stream>>>(aw1, atB, 1024, 32, 8);

  // Encoder
  conv_pool_relu_direct<3, 8, 128><<<512 * 8 * 16, 256, 0, stream>>>(data, cw1, cb1, p1);
  conv_pool_relu_direct<8, 16, 64><<<512 * 16 * 4, 256, 0, stream>>>(p1, cw2, cb2, p2);
  conv_pool_relu_wmma<16, 32, 32><<<512, 256, 0, stream>>>(p2, wsw3, cb3, p3);
  conv_pool_relu_wmma<32, 64, 16><<<512, 256, 0, stream>>>(p3, wsw4, cb4, p4);
  conv_pool_relu_wmma<64, 128, 8><<<512, 256, 0, stream>>>(p4, wsw5, cb5, p5);

  // FC: emb(512x1024) = flat(512x2048) @ fc_w^T + fc_b
  swizzle_a_bf16<<<gs(32LL * 64 * 32), 256, 0, stream>>>(p5, fcA, 2048, 64, 32);
  gemm_wmma<<<(32 * 64) / 8, 256, 0, stream>>>(fcA, fcB, fcb, emb, 64, 64, 1024, 32 * 64, 0);

  // Attention hidden: h(512x128) = tanh(emb @ att_w1^T + att_b1)
  swizzle_a<<<gs(32LL * 32 * 32), 256, 0, stream>>>(emb, atA, 1024, 32, 32);
  gemm_wmma<<<(32 * 8) / 8, 256, 0, stream>>>(atA, atB, ab1, hatt, 8, 32, 128, 32 * 8, 1);

  // Tail
  att_logits<<<2, 256, 0, stream>>>(hatt, aw2, ab2, lgt);
  segment_attn<<<64, 256, 0, stream>>>(lgt, ids, emb, aggr);
  classify<<<1, 64, 0, stream>>>(aggr, clw, clb, outp);
}
